// CrissCrossAttention_6073083757210
// MI455X (gfx1250) — compile-verified, matrix operand order checked by reading, not run on hardware
//
#include <hip/hip_runtime.h>
#include <hip/hip_bf16.h>

#define B_ 16
#define H_ 128
#define W_ 128
#define C_ 256
#define D_ 32
#define M_ (B_*H_*W_)          // 262144 queries
#define NQKV_ 320              // 32(q)+32(k)+256(v)
#define GAMMA_ 0.05f

typedef __attribute__((ext_vector_type(16))) __bf16 v16bf;
typedef __attribute__((ext_vector_type(8)))  float  v8f;

// ---- async global->LDS staging (gfx1250), guarded -------------------------
#if defined(__gfx1250__) && __has_builtin(__builtin_amdgcn_global_load_async_to_lds_b32)
#define ASYNC_LDS 1
typedef __attribute__((address_space(1))) int as1_int;
typedef __attribute__((address_space(3))) int as3_int;
#endif

__device__ __forceinline__ void stage_b32(const void* g, void* l) {
#ifdef ASYNC_LDS
  __builtin_amdgcn_global_load_async_to_lds_b32((as1_int*)g, (as3_int*)l, 0, 0);
#else
  *(unsigned int*)l = *(const unsigned int*)g;
#endif
}
__device__ __forceinline__ void stage_wait() {
#ifdef ASYNC_LDS
  asm volatile("s_wait_asynccnt 0x0" ::: "memory");
#endif
}

// ---- WMMA operand loaders -------------------------------------------------
// A-matrix 16x32 bf16 per-lane layout (ISA 7.12.2):
//   lanes 0-15  (hi=0): K = {0..7, 16..23};  lanes 16-31 (hi=1): K = {8..15, 24..31}
// dword d: K0 = d<4 ? hi*8+2d : 16+hi*8+2(d-4)  (two contiguous b128 per operand)
__device__ __forceinline__ int kpat_dw(int d, int hi) {
  return (d < 4) ? (hi * 8 + 2 * d) : (16 + hi * 8 + 2 * (d - 4));
}

template <typename PT>
__device__ __forceinline__ v16bf load_k32(const PT* row, int k0, int hi) {
  union { unsigned int u[8]; v16bf v; } t;
#pragma unroll
  for (int d = 0; d < 8; ++d)
    t.u[d] = *reinterpret_cast<const unsigned int*>(row + (k0 + kpat_dw(d, hi)));
  return t.v;
}

// A-operand directly from fp32 memory with on-the-fly bf16 conversion.
__device__ __forceinline__ v16bf cvt_k32_f32(const float* row, int k0, int hi) {
  float4 f0 = *(const float4*)(row + k0 + hi * 8);
  float4 f1 = *(const float4*)(row + k0 + hi * 8 + 4);
  float4 f2 = *(const float4*)(row + k0 + 16 + hi * 8);
  float4 f3 = *(const float4*)(row + k0 + 16 + hi * 8 + 4);
  union { __bf16 h[16]; v16bf v; } t;
  t.h[0] = (__bf16)f0.x;  t.h[1] = (__bf16)f0.y;
  t.h[2] = (__bf16)f0.z;  t.h[3] = (__bf16)f0.w;
  t.h[4] = (__bf16)f1.x;  t.h[5] = (__bf16)f1.y;
  t.h[6] = (__bf16)f1.z;  t.h[7] = (__bf16)f1.w;
  t.h[8] = (__bf16)f2.x;  t.h[9] = (__bf16)f2.y;
  t.h[10] = (__bf16)f2.z; t.h[11] = (__bf16)f2.w;
  t.h[12] = (__bf16)f3.x; t.h[13] = (__bf16)f3.y;
  t.h[14] = (__bf16)f3.z; t.h[15] = (__bf16)f3.w;
  return t.v;
}

__device__ __forceinline__ v8f wmma_bf16(v16bf a, v16bf b, v8f c) {
  return __builtin_amdgcn_wmma_f32_16x16x32_bf16(false, a, false, b, (short)0, c,
                                                 false, false);
}

// ---- Kernel A: pack [wq|wk|wv] transposed to bf16 WcT[n][c], bias bc[n] ----
__global__ void cc_pack_w(const float* wq, const float* wk, const float* wv,
                          const float* bq, const float* bk, const float* bv,
                          __bf16* WcT, float* bc) {
  int t = blockIdx.x * blockDim.x + threadIdx.x;
  if (t < NQKV_ * C_) {
    int n = t / C_, c = t % C_;
    float v;
    if (n < 32)       v = wq[c * 32 + n];
    else if (n < 64)  v = wk[c * 32 + (n - 32)];
    else              v = wv[c * C_ + (n - 64)];
    WcT[n * C_ + c] = (__bf16)v;
  }
  if (t < NQKV_)
    bc[t] = (t < 32) ? bq[t] : ((t < 64) ? bk[t - 32] : bv[t - 64]);
}

// ---- Kernel B: fused QKV projection, x fp32 read + inline bf16 convert ----
__global__ void __launch_bounds__(256, 1)
cc_qkv(const float* __restrict__ x, const __bf16* __restrict__ WcT,
       const float* __restrict__ bc, __bf16* __restrict__ qb,
       __bf16* __restrict__ kb, __bf16* __restrict__ vb) {
  const int tid = threadIdx.x, lane = tid & 31, wvid = tid >> 5;
  const int l16 = lane & 15, hi = lane >> 4;
  const long m0 = (long)blockIdx.x * 128 + wvid * 16;
  const float* xrow = x + (m0 + l16) * C_;

  v16bf a[8];                          // 16x256 A strip resident in VGPRs
#pragma unroll
  for (int ks = 0; ks < 8; ++ks) a[ks] = cvt_k32_f32(xrow, ks * 32, hi);

#pragma unroll 1
  for (int nt = 0; nt < 20; ++nt) {
    const int n = nt * 16 + l16;
    v8f acc = {};
#pragma unroll
    for (int ks = 0; ks < 8; ++ks) {
      v16bf bm = load_k32(WcT + (long)n * C_, ks * 32, hi);
      acc = wmma_bf16(a[ks], bm, acc);
    }
    const float bias = bc[n];
    __bf16* dst; long stride; int nn;          // uniform per tile (nt)
    if (nt < 2)      { dst = qb; stride = D_; nn = n; }
    else if (nt < 4) { dst = kb; stride = D_; nn = n - 32; }
    else             { dst = vb; stride = C_; nn = n - 64; }
#pragma unroll
    for (int r = 0; r < 8; ++r)
      dst[(m0 + r + hi * 8) * stride + nn] = (__bf16)(acc[r] + bias);
  }
}

// LDS carve (elems): Q 128x40, K 128x40, VT 256x136, S 128x132 f32, P 128x136
#define QS_STR 40
#define VT_STR 136
#define S_STR  132
#define P_STR  136

// ---- Kernel C: column attention pass (per (b,w)) --------------------------
__global__ void __launch_bounds__(256, 1)
cc_col(const __bf16* __restrict__ qb, const __bf16* __restrict__ kb,
       const __bf16* __restrict__ vb, __bf16* __restrict__ Oh,
       float* __restrict__ mh, float* __restrict__ sh) {
  extern __shared__ char smem[];
  __bf16* Qs = (__bf16*)smem;
  __bf16* Ks = Qs + 128 * QS_STR;
  __bf16* VT = Ks + 128 * QS_STR;
  float*  S  = (float*)(VT + 256 * VT_STR);
  __bf16* P  = (__bf16*)(S + 128 * S_STR);

  const int tid = threadIdx.x, lane = tid & 31, wvid = tid >> 5;
  const int l16 = lane & 15, hi = lane >> 4;
  const int b = blockIdx.x / W_, w = blockIdx.x % W_;

  for (int i = tid; i < 128 * 16; i += 256) {         // Q,K column strips
    int row = i >> 4, dw = i & 15;
    long off = ((long)(b * H_ + row) * W_ + w) * D_ + dw * 2;
    stage_b32(qb + off, Qs + row * QS_STR + dw * 2);
    stage_b32(kb + off, Ks + row * QS_STR + dw * 2);
  }
  for (int i = tid; i < 128 * 128; i += 256) {        // V column, transposed
    int row = i >> 7, dwc = i & 127;
    long off = ((long)(b * H_ + row) * W_ + w) * C_ + dwc * 2;
    unsigned int val = *(const unsigned int*)(vb + off);
    VT[(dwc * 2 + 0) * VT_STR + row] = ((const __bf16*)&val)[0];
    VT[(dwc * 2 + 1) * VT_STR + row] = ((const __bf16*)&val)[1];
  }
  stage_wait();
  __syncthreads();

  const int m0 = wvid * 16;                           // S = Q K^T (K-dim = 32)
  v16bf a = load_k32(Qs + (m0 + l16) * QS_STR, 0, hi);
#pragma unroll
  for (int t = 0; t < 8; ++t) {
    v16bf bm = load_k32(Ks + (t * 16 + l16) * QS_STR, 0, hi);
    v8f acc = {};
    acc = wmma_bf16(a, bm, acc);
    int n = t * 16 + l16;
#pragma unroll
    for (int r = 0; r < 8; ++r) {
      int m = m0 + r + hi * 8;
      S[m * S_STR + n] = (n == m) ? -1e30f : acc[r];  // mask diagonal (k==h)
    }
  }
  __syncthreads();

  if (tid < 128) {                                    // row softmax (partial)
    float mx = -1e30f;
    for (int j = 0; j < 128; ++j) mx = fmaxf(mx, S[tid * S_STR + j]);
    float sum = 0.f;
    for (int j = 0; j < 128; ++j) {
      float p = __expf(S[tid * S_STR + j] - mx);
      sum += p;
      P[tid * P_STR + j] = (__bf16)p;
    }
    long qidx = (long)(b * H_ + tid) * W_ + w;
    mh[qidx] = mx; sh[qidx] = sum;
  }
  __syncthreads();

#pragma unroll 1
  for (int ct = 0; ct < 16; ++ct) {                   // O_h = P @ V
    v8f acc = {};
#pragma unroll
    for (int ks = 0; ks < 4; ++ks) {
      v16bf pa = load_k32(P + (m0 + l16) * P_STR, ks * 32, hi);
      v16bf vm = load_k32(VT + (ct * 16 + l16) * VT_STR, ks * 32, hi);
      acc = wmma_bf16(pa, vm, acc);
    }
    int n = ct * 16 + l16;
#pragma unroll
    for (int r = 0; r < 8; ++r) {
      int m = m0 + r + hi * 8;
      long qidx = (long)(b * H_ + m) * W_ + w;
      Oh[qidx * C_ + n] = (__bf16)acc[r];
    }
  }
}

// ---- Kernel D: row attention pass + softmax merge + epilogue --------------
__global__ void __launch_bounds__(256, 1)
cc_row(const __bf16* __restrict__ qb, const __bf16* __restrict__ kb,
       const __bf16* __restrict__ vb, const __bf16* __restrict__ Oh,
       const float* __restrict__ mh, const float* __restrict__ sh,
       const float* __restrict__ x, float* __restrict__ out) {
  extern __shared__ char smem[];
  __bf16* Qs = (__bf16*)smem;
  __bf16* Ks = Qs + 128 * QS_STR;
  __bf16* VT = Ks + 128 * QS_STR;
  float*  S  = (float*)(VT + 256 * VT_STR);
  __bf16* P  = (__bf16*)(S + 128 * S_STR);
  float*  Mw = (float*)(P + 128 * P_STR);
  float*  Sw = Mw + 128;
  float*  Mh = Sw + 128;
  float*  Sh = Mh + 128;

  const int tid = threadIdx.x, lane = tid & 31, wvid = tid >> 5;
  const int l16 = lane & 15, hi = lane >> 4;
  const int b = blockIdx.x / H_, h = blockIdx.x % H_;
  const __bf16* qbase = qb + (long)(b * H_ + h) * W_ * D_;
  const __bf16* kbase = kb + (long)(b * H_ + h) * W_ * D_;
  const __bf16* vbase = vb + (long)(b * H_ + h) * W_ * C_;

  for (int i = tid; i < 128 * 16; i += 256) {         // contiguous row strips
    int row = i >> 4, dw = i & 15;
    stage_b32(qbase + row * D_ + dw * 2, Qs + row * QS_STR + dw * 2);
    stage_b32(kbase + row * D_ + dw * 2, Ks + row * QS_STR + dw * 2);
  }
  for (int i = tid; i < 128 * 128; i += 256) {
    int row = i >> 7, dwc = i & 127;
    unsigned int val = *(const unsigned int*)(vbase + row * C_ + dwc * 2);
    VT[(dwc * 2 + 0) * VT_STR + row] = ((const __bf16*)&val)[0];
    VT[(dwc * 2 + 1) * VT_STR + row] = ((const __bf16*)&val)[1];
  }
  stage_wait();
  __syncthreads();

  const int m0 = wvid * 16;
  v16bf a = load_k32(Qs + (m0 + l16) * QS_STR, 0, hi);
#pragma unroll
  for (int t = 0; t < 8; ++t) {
    v16bf bm = load_k32(Ks + (t * 16 + l16) * QS_STR, 0, hi);
    v8f acc = {};
    acc = wmma_bf16(a, bm, acc);
    int n = t * 16 + l16;
#pragma unroll
    for (int r = 0; r < 8; ++r) S[(m0 + r + hi * 8) * S_STR + n] = acc[r]; // no mask
  }
  __syncthreads();

  if (tid < 128) {
    float mx = -1e30f;
    for (int j = 0; j < 128; ++j) mx = fmaxf(mx, S[tid * S_STR + j]);
    float sum = 0.f;
    for (int j = 0; j < 128; ++j) {
      float p = __expf(S[tid * S_STR + j] - mx);
      sum += p;
      P[tid * P_STR + j] = (__bf16)p;
    }
    Mw[tid] = mx; Sw[tid] = sum;
    long qidx = (long)(b * H_ + h) * W_ + tid;
    Mh[tid] = mh[qidx]; Sh[tid] = sh[qidx];
  }
  __syncthreads();

#pragma unroll 1
  for (int ct = 0; ct < 16; ++ct) {                   // O_w = P @ V, then merge
    v8f acc = {};
#pragma unroll
    for (int ks = 0; ks < 4; ++ks) {
      v16bf pa = load_k32(P + (m0 + l16) * P_STR, ks * 32, hi);
      v16bf vm = load_k32(VT + (ct * 16 + l16) * VT_STR, ks * 32, hi);
      acc = wmma_bf16(pa, vm, acc);
    }
    int n = ct * 16 + l16;
#pragma unroll
    for (int r = 0; r < 8; ++r) {
      int m = m0 + r + hi * 8;                        // query's w coordinate
      float mhv = Mh[m], shv = Sh[m], mwv = Mw[m], swv = Sw[m];
      float mm = fmaxf(mhv, mwv);
      float ah = __expf(mhv - mm), aw = __expf(mwv - mm);
      float denom = shv * ah + swv * aw;
      long qidx = (long)(b * H_ + h) * W_ + m;
      float oh = (float)Oh[qidx * C_ + n];
      float xv = x[qidx * C_ + n];
      out[qidx * C_ + n] = GAMMA_ * ((oh * ah + acc[r] * aw) / denom) + xv;
    }
  }
}

// ---- host launch ----------------------------------------------------------
extern "C" void kernel_launch(void* const* d_in, const int* in_sizes, int n_in,
                              void* d_out, int out_size, void* d_ws, size_t ws_size,
                              hipStream_t stream) {
  const float* x  = (const float*)d_in[0];
  const float* wq = (const float*)d_in[1];
  const float* bq = (const float*)d_in[2];
  const float* wk = (const float*)d_in[3];
  const float* bk = (const float*)d_in[4];
  const float* wv = (const float*)d_in[5];
  const float* bv = (const float*)d_in[6];
  float* out = (float*)d_out;

  char* p = (char*)d_ws;
  size_t off = 0;
  auto carve = [&](size_t bytes) {
    void* r = p + off;
    off = (off + bytes + 255) & ~(size_t)255;
    return r;
  };
  __bf16* qb  = (__bf16*)carve((size_t)M_ * D_ * 2);
  __bf16* kb  = (__bf16*)carve((size_t)M_ * D_ * 2);
  __bf16* vb  = (__bf16*)carve((size_t)M_ * C_ * 2);
  __bf16* WcT = (__bf16*)carve((size_t)NQKV_ * C_ * 2);
  float*  bc  = (float*)carve((size_t)NQKV_ * 4);
  __bf16* Oh  = (__bf16*)carve((size_t)M_ * C_ * 2);
  float*  mh  = (float*)carve((size_t)M_ * 4);
  float*  sh  = (float*)carve((size_t)M_ * 4);
  (void)ws_size; (void)in_sizes; (void)n_in; (void)out_size;

  const size_t smem_col = (size_t)(128 * QS_STR * 2) * 2 + 256 * VT_STR * 2 +
                          128 * S_STR * 4 + 128 * P_STR * 2;
  const size_t smem_row = smem_col + 4 * 128 * 4;

  cc_pack_w<<<(NQKV_ * C_ + 255) / 256, 256, 0, stream>>>(wq, wk, wv, bq, bk, bv,
                                                          WcT, bc);
  cc_qkv<<<M_ / 128, 256, 0, stream>>>(x, WcT, bc, qb, kb, vb);
  cc_col<<<B_ * W_, 256, smem_col, stream>>>(qb, kb, vb, Oh, mh, sh);
  cc_row<<<B_ * H_, 256, smem_row, stream>>>(qb, kb, vb, Oh, mh, sh, x, out);
}
